// PPNUResNet_10333691314301
// MI455X (gfx1250) — matrix operand order repaired and checked by reading, not compiled, not run on hardware
//
#include <hip/hip_runtime.h>

// ---------------------------------------------------------------------------
// Types for CDNA5 WMMA (wave32): V_WMMA_F32_16X16X32_BF16
// ---------------------------------------------------------------------------
typedef __attribute__((ext_vector_type(16))) __bf16 v16bf;
typedef __attribute__((ext_vector_type(8)))  float  v8f;
typedef __attribute__((ext_vector_type(4)))  unsigned u32x4;
typedef __attribute__((ext_vector_type(8)))  int      i32x8;
typedef __attribute__((ext_vector_type(4)))  int      i32x4;

// ---------------------------------------------------------------------------
// Weight repack: DHWIO fp32 -> bf16 tiles [coutTile*16][KtotPad], zero padded.
// grid = (ceil(KtotPad/256), CoutPad) : no divisions.
// ---------------------------------------------------------------------------
__global__ void repack_kernel(const float* __restrict__ w, __bf16* __restrict__ wbf,
                              int Ktot, int KtotPad, int Cout) {
  const int nc = blockIdx.y;
  for (int k = blockIdx.x * 256 + threadIdx.x; k < KtotPad; k += gridDim.x * 256) {
    float v = 0.0f;
    if (k < Ktot && nc < Cout) v = w[(long)k * Cout + nc];
    wbf[(long)nc * KtotPad + k] = (__bf16)v;
  }
}

// ---------------------------------------------------------------------------
// Fast implicit-GEMM conv3d (Cin multiple of 16), NDHWC, bf16 WMMA, f32 acc.
//  - 8 waves/block share one 16-out-channel tile
//  - B tile (16 x KtotPad bf16) staged into LDS via the Tensor Data Mover
//  - A gathered from global as two float4 runs per 32-K chunk; the K->(tap,c)
//    mapping is carried incrementally (no divisions in the loop)
//  mode 0: dense conv kd^3/stride/pad ; mode 1: transposed 2^3 stride-2
// ---------------------------------------------------------------------------
__global__ __launch_bounds__(256) void conv_fast_kernel(
    const float* __restrict__ in, const __bf16* __restrict__ wbf,
    const float* __restrict__ bias, float* __restrict__ out,
    int s_in, int lgSin, int s_out, int lgSout,
    int Cin, int Cout, int kd, int stride, int pad, int mode,
    int Ktot, int KtotPad)
{
  extern __shared__ char smem_raw[];
  __bf16* sB = reinterpret_cast<__bf16*>(smem_raw);
  const int ct = blockIdx.y;
  const int tileElems = KtotPad << 4;   // 16 rows

#if defined(__has_builtin) && __has_builtin(__builtin_amdgcn_tensor_load_to_lds) && \
    __has_builtin(__builtin_amdgcn_s_wait_tensorcnt)
  if ((threadIdx.x >> 5) == 0) {
    const unsigned long long gaddr =
        (unsigned long long)(uintptr_t)(wbf + (size_t)ct * tileElems);
    typedef __attribute__((address_space(3))) char lds_char;
    const unsigned ldsOff = (unsigned)(uintptr_t)(lds_char*)smem_raw;
    u32x4 g0 = { 1u,                                    // count=1, user desc
                 ldsOff,                                // lds_addr
                 (unsigned)gaddr,                       // global_addr[31:0]
                 (unsigned)((gaddr >> 32) & 0x1FFFFFFu) | (2u << 30) }; // [56:32]|type=2
    const unsigned KP = (unsigned)KtotPad;
    i32x8 g1 = { (int)(1u << 16),                       // data_size = 2 bytes
                 (int)((KP & 0xFFFFu) << 16),           // tensor_dim0[15:0]
                 (int)((KP >> 16) | (16u << 16)),       // tensor_dim0[31:16] | tensor_dim1
                 (int)(KP << 16),                       // tile_dim0
                 16,                                    // tile_dim1 (rows)
                 (int)KP,                               // tensor_dim0_stride[31:0]
                 0, 0 };
    i32x4 z4 = { 0, 0, 0, 0 };
#if __clang_major__ >= 23
    i32x8 z8 = { 0, 0, 0, 0, 0, 0, 0, 0 };
    __builtin_amdgcn_tensor_load_to_lds(g0, g1, z4, z4, z8, 0);
#else
    __builtin_amdgcn_tensor_load_to_lds(g0, g1, z4, z4, 0);
#endif
    __builtin_amdgcn_s_wait_tensorcnt(0);
  }
#else
  {
    const __bf16* src = wbf + (size_t)ct * tileElems;
    for (int i = threadIdx.x; i < tileElems; i += 256) sB[i] = src[i];
  }
#endif
  __syncthreads();

  const int nVoxT = (s_out * s_out * s_out) >> 4;
  const int vt = blockIdx.x * 8 + (threadIdx.x >> 5);
  if (vt >= nVoxT) return;

  const int lane = threadIdx.x & 31;
  const int lh = lane >> 4, ll = lane & 15;
  const int vox = (vt << 4) + ll;
  const int sm = s_out - 1;
  const int x = vox & sm;
  const int y = (vox >> lgSout) & sm;
  const int z = vox >> (lgSout << 1);
  const int nB = (ct << 4) + ll;
  const int taps = (mode == 1) ? 8 : kd * kd * kd;
  const int tpar = ((z & 1) << 2) | ((y & 1) << 1) | (x & 1);
  const int zb = (mode == 1) ? (z >> 1) : z * stride - pad;
  const int yb = (mode == 1) ? (y >> 1) : y * stride - pad;
  const int xb = (mode == 1) ? (x >> 1) : x * stride - pad;

  // per-run incremental K state (run k = k0 + lh*16 + r*8; 8-runs never straddle taps)
  int tapR[2], cR[2], dxR[2], dyR[2], dzR[2];
#pragma unroll
  for (int r = 0; r < 2; ++r) {
    const int kinit = (lh << 4) + (r << 3);
    const int tp = (kinit >= Cin) ? 1 : 0;
    tapR[r] = tp; cR[r] = kinit - (tp ? Cin : 0);
    dxR[r] = tp; dyR[r] = 0; dzR[r] = 0;
  }

  __builtin_prefetch(in, 0, 3);

  v8f acc = {};
#pragma unroll 2
  for (int k0 = 0; k0 < Ktot; k0 += 32) {
    v16bf a;
#pragma unroll
    for (int r = 0; r < 2; ++r) {
      float4 lo = make_float4(0.f, 0.f, 0.f, 0.f);
      float4 hi = make_float4(0.f, 0.f, 0.f, 0.f);
      bool ok = (tapR[r] < taps);
      int iz, iy, ix;
      if (mode == 1) {
        ok = ok && (tapR[r] == tpar);
        iz = zb; iy = yb; ix = xb;
      } else {
        iz = zb + dzR[r]; iy = yb + dyR[r]; ix = xb + dxR[r];
        ok = ok & ((unsigned)iz < (unsigned)s_in) &
                  ((unsigned)iy < (unsigned)s_in) &
                  ((unsigned)ix < (unsigned)s_in);
      }
      if (ok) {
        const float4* p = reinterpret_cast<const float4*>(
            in + (long)((((iz << lgSin) | iy) << lgSin) | ix) * Cin + cR[r]);
        lo = p[0]; hi = p[1];
      }
      const int j0 = r << 3;
      a[j0 + 0] = (__bf16)lo.x; a[j0 + 1] = (__bf16)lo.y;
      a[j0 + 2] = (__bf16)lo.z; a[j0 + 3] = (__bf16)lo.w;
      a[j0 + 4] = (__bf16)hi.x; a[j0 + 5] = (__bf16)hi.y;
      a[j0 + 6] = (__bf16)hi.z; a[j0 + 7] = (__bf16)hi.w;
      // advance this run's K by 32 (at most two tap wraps since Cin >= 16)
      cR[r] += 32;
      if (cR[r] >= Cin) {
        cR[r] -= Cin; tapR[r] += 1;
        if (++dxR[r] == kd) { dxR[r] = 0; if (++dyR[r] == kd) { dyR[r] = 0; ++dzR[r]; } }
        if (cR[r] >= Cin) {
          cR[r] -= Cin; tapR[r] += 1;
          if (++dxR[r] == kd) { dxR[r] = 0; if (++dyR[r] == kd) { dyR[r] = 0; ++dzR[r]; } }
        }
      }
    }
    const v16bf b = *reinterpret_cast<const v16bf*>(&sB[ll * KtotPad + k0 + (lh << 4)]);
    acc = __builtin_amdgcn_wmma_f32_16x16x32_bf16(
        false, a, false, b, (short)0, acc, false, false);
  }

  if (nB < Cout) {
    const float bb = bias ? bias[nB] : 0.0f;
#pragma unroll
    for (int r = 0; r < 8; ++r) {
      const int vo = (vt << 4) + r + (lh << 3);
      out[(long)vo * Cout + nB] = acc[r] + bb;
    }
  }
}

// ---------------------------------------------------------------------------
// Input conv (Cin==1, 3^3 SAME, s=64, Cout=16): Ktot=27 -> single WMMA.
// ---------------------------------------------------------------------------
__global__ __launch_bounds__(256) void conv_c1_kernel(
    const float* __restrict__ in, const float* __restrict__ w,
    float* __restrict__ out)
{
  const int s = 64;
  const int vt = blockIdx.x * 8 + (threadIdx.x >> 5);
  const int lane = threadIdx.x & 31;
  const int lh = lane >> 4, ll = lane & 15;
  const int vox = (vt << 4) + ll;
  const int x = vox & 63, y = (vox >> 6) & 63, z = vox >> 12;

  v16bf a, b;
#pragma unroll
  for (int j = 0; j < 16; ++j) {
    const int vg = j >> 1, par = j & 1;
    const int ka = ((vg < 4) ? 0 : 16) + (lh << 3) + ((vg & 3) << 1) + par;
    float av = 0.0f;
    if (ka < 27) {
      const int dz = ka / 9; const int rr = ka - dz * 9;
      const int dy = rr / 3; const int dx = rr - dy * 3;
      const int iz = z + dz - 1, iy = y + dy - 1, ix = x + dx - 1;
      if (((unsigned)iz < 64u) & ((unsigned)iy < 64u) & ((unsigned)ix < 64u))
        av = in[((iz << 6) | iy) << 6 | ix];
    }
    a[j] = (__bf16)av;
    const int kb = ka;
    b[j] = (kb < 27) ? (__bf16)w[(kb << 4) + ll] : (__bf16)0.0f;
  }
  v8f acc = {};
  acc = __builtin_amdgcn_wmma_f32_16x16x32_bf16(
      false, a, false, b, (short)0, acc, false, false);
#pragma unroll
  for (int r = 0; r < 8; ++r) {
    const int vo = (vt << 4) + r + (lh << 3);
    out[(vo << 4) + ll] = acc[r];
  }
}

// ---------------------------------------------------------------------------
// Elementwise / reduction helper kernels
// (grids for channel-indexed kernels are multiples of C so c is loop-invariant)
// ---------------------------------------------------------------------------
__global__ void zero_kernel(float* p, long n) {
  for (long i = blockIdx.x * (long)blockDim.x + threadIdx.x; i < n;
       i += (long)gridDim.x * blockDim.x) p[i] = 0.0f;
}
__global__ void copy_kernel(const float* __restrict__ a, float* __restrict__ o, long n) {
  for (long i = blockIdx.x * (long)blockDim.x + threadIdx.x; i < n;
       i += (long)gridDim.x * blockDim.x) o[i] = a[i];
}
__global__ void add_kernel(const float* __restrict__ a, const float* __restrict__ b,
                           float* __restrict__ o, long n) {
  for (long i = blockIdx.x * (long)blockDim.x + threadIdx.x; i < n;
       i += (long)gridDim.x * blockDim.x) o[i] = a[i] + b[i];
}
__global__ void concat_kernel(const float* __restrict__ a, int Ca,
                              const float* __restrict__ b, int Cb,
                              float* __restrict__ o, long nvox) {
  const int C = Ca + Cb;
  const long n = nvox * C;
  const long stride = (long)gridDim.x * blockDim.x;   // multiple of C
  long i = blockIdx.x * (long)blockDim.x + threadIdx.x;
  if (i >= n) return;
  const int c = (int)(i % C);
  long v = i / C;
  const long vstep = stride / C;
  const float* src = (c < Ca) ? (a + c) : (b + (c - Ca));
  const int cs = (c < Ca) ? Ca : Cb;
  for (; i < n; i += stride, v += vstep) o[i] = src[v * cs];
}
__global__ void bn_stats_kernel(const float* __restrict__ x, long nvox, int C,
                                float* __restrict__ st) {
  const long n = nvox * C;
  const long stride = (long)gridDim.x * blockDim.x;   // multiple of C
  long i = blockIdx.x * (long)blockDim.x + threadIdx.x;
  const int c = (int)(i % C);
  float s0 = 0.f, s1 = 0.f;
  for (; i < n; i += stride) { const float v = x[i]; s0 += v; s1 += v * v; }
  atomicAdd(&st[c], s0);
  atomicAdd(&st[C + c], s1);
}
__global__ void bn_apply_kernel(const float* __restrict__ x, float* __restrict__ y,
                                long nvox, int C, const float* __restrict__ st,
                                const float* __restrict__ g, const float* __restrict__ b) {
  const long n = nvox * C;
  const long stride = (long)gridDim.x * blockDim.x;   // multiple of C
  long i = blockIdx.x * (long)blockDim.x + threadIdx.x;
  if (i >= n) return;
  const int c = (int)(i % C);
  const float inv = 1.0f / (float)nvox;
  const float mu  = st[c] * inv;
  const float var = st[C + c] * inv - mu * mu;
  const float sc  = rsqrtf(var + 1e-4f) * g[c];
  const float sh  = b[c] - mu * sc;
  for (; i < n; i += stride) {
    const float v = fmaf(x[i], sc, sh);
    y[i] = v > 0.0f ? v : 0.0f;
  }
}
__global__ void selmask_kernel(const float* __restrict__ sc, float* __restrict__ mask, long n) {
  for (long i = blockIdx.x * (long)blockDim.x + threadIdx.x; i < n;
       i += (long)gridDim.x * blockDim.x) {
    const float a = sc[2 * i], b = sc[2 * i + 1];
    const float m = fmaxf(a, b);
    const float e0 = __expf(a - m), e1 = __expf(b - m);
    mask[i] = (e1 / (e0 + e1) > 0.8f) ? 1.0f : 0.0f;
  }
}
__global__ void maskmul_kernel(const float* __restrict__ feat, const float* __restrict__ mask,
                               float* __restrict__ out, int lgs, int lgf, int C) {
  const int s = 1 << lgs, sm = s - 1, lgscs = lgs - lgf;
  const long n = ((long)1 << (3 * lgs)) * C;
  const long stride = (long)gridDim.x * blockDim.x;   // multiple of C
  long i = blockIdx.x * (long)blockDim.x + threadIdx.x;
  if (i >= n) return;
  long v = i / C;
  const long vstep = stride / C;
  for (; i < n; i += stride, v += vstep) {
    const int x = (int)v & sm;
    const int y = ((int)(v >> lgs)) & sm;
    const int z = (int)(v >> (lgs << 1));
    const long ci = ((((long)(z >> lgf)) << lgscs | (y >> lgf)) << lgscs) | (x >> lgf);
    out[i] = feat[i] * mask[ci];
  }
}
__global__ void attwrite_kernel(const float* __restrict__ mask, int lgscs, int lgf,
                                float* __restrict__ dst) {
  const int lgs = lgscs + lgf;
  const int sm = (1 << lgs) - 1;
  const long n = (long)1 << (3 * lgs);
  for (long i = blockIdx.x * (long)blockDim.x + threadIdx.x; i < n;
       i += (long)gridDim.x * blockDim.x) {
    const int x = (int)i & sm;
    const int y = ((int)(i >> lgs)) & sm;
    const int z = (int)(i >> (lgs << 1));
    dst[i] = mask[((((long)(z >> lgf)) << lgscs | (y >> lgf)) << lgscs) | (x >> lgf)];
  }
}
__global__ void points_kernel(const float* __restrict__ pp, const float* __restrict__ sc,
                              float* __restrict__ dst, long n) {
  for (long i = blockIdx.x * (long)blockDim.x + threadIdx.x; i < n;
       i += (long)gridDim.x * blockDim.x) {
    dst[i * 5 + 0] = pp[i * 3 + 0];
    dst[i * 5 + 1] = pp[i * 3 + 1];
    dst[i * 5 + 2] = pp[i * 3 + 2];
    dst[i * 5 + 3] = sc[i * 2 + 0];
    dst[i * 5 + 4] = sc[i * 2 + 1];
  }
}
__global__ void ppnwrite_kernel(const float* __restrict__ sc, float* __restrict__ dst, int lgs) {
  const int sm = (1 << lgs) - 1;
  const long n = (long)1 << (3 * lgs);
  for (long i = blockIdx.x * (long)blockDim.x + threadIdx.x; i < n;
       i += (long)gridDim.x * blockDim.x) {
    dst[i * 6 + 0] = (float)((int)(i >> (lgs << 1)));
    dst[i * 6 + 1] = (float)(((int)(i >> lgs)) & sm);
    dst[i * 6 + 2] = (float)((int)i & sm);
    dst[i * 6 + 3] = 0.0f;
    dst[i * 6 + 4] = sc[i * 2 + 0];
    dst[i * 6 + 5] = sc[i * 2 + 1];
  }
}

// ---------------------------------------------------------------------------
// Host orchestration
// ---------------------------------------------------------------------------
struct BlkP { const float *b1, *b2, *g1, *g2, *nin, *w1, *w2; };

extern "C" void kernel_launch(void* const* d_in, const int* in_sizes, int n_in,
                              void* d_out, int out_size, void* d_ws, size_t ws_size,
                              hipStream_t stream) {
  (void)in_sizes; (void)out_size; (void)ws_size;
  if (n_in < 149) return;

  // ---- pytree-flat input order (jax: dicts sorted by key, None dropped) ----
  int cidx = 0;
  auto nxt = [&]() { return (const float*)d_in[cidx++]; };

  struct DecP { const float* b; BlkP blk[2]; const float* g; const float* w; } dec[4];
  for (int j = 0; j < 4; ++j) {
    dec[j].b = nxt();
    dec[j].blk[0].b1 = nxt(); dec[j].blk[0].b2 = nxt();
    dec[j].blk[0].g1 = nxt(); dec[j].blk[0].g2 = nxt();
    dec[j].blk[0].nin = nxt();
    dec[j].blk[0].w1 = nxt(); dec[j].blk[0].w2 = nxt();
    dec[j].blk[1].b1 = nxt(); dec[j].blk[1].b2 = nxt();
    dec[j].blk[1].g1 = nxt(); dec[j].blk[1].g2 = nxt();
    dec[j].blk[1].nin = nullptr;
    dec[j].blk[1].w1 = nxt(); dec[j].blk[1].w2 = nxt();
    dec[j].g = nxt(); dec[j].w = nxt();
  }
  struct EncP { BlkP blk[2]; const float *db, *dg, *dw; } enc[5];
  for (int i = 0; i < 5; ++i) {
    for (int bi = 0; bi < 2; ++bi) {
      enc[i].blk[bi].b1 = nxt(); enc[i].blk[bi].b2 = nxt();
      enc[i].blk[bi].g1 = nxt(); enc[i].blk[bi].g2 = nxt();
      enc[i].blk[bi].nin = nullptr;
      enc[i].blk[bi].w1 = nxt(); enc[i].blk[bi].w2 = nxt();
    }
    if (i < 4) { enc[i].db = nxt(); enc[i].dg = nxt(); enc[i].dw = nxt(); }
    else { enc[i].db = enc[i].dg = enc[i].dw = nullptr; }
  }
  const float* out_b = nxt(); const float* out_blin = nxt();
  const float* out_g = nxt(); const float* out_wlin = nxt();
  const float* w1c = nxt(); const float* w1s = nxt();
  const float* w2c = nxt(); const float* w2s = nxt();
  const float* w3c = nxt(); const float* w3p = nxt(); const float* w3s = nxt();
  const float* w_in = nxt();
  const float* x_in = nxt();

  // ---- workspace bump allocator ----
  size_t cur = 0;
  auto alloc = [&](long nfloats) {
    float* p = (float*)((char*)d_ws + cur);
    cur = (cur + (size_t)nfloats * sizeof(float) + 255) & ~(size_t)255;
    return p;
  };
  const long N64 = 64L * 64 * 64;
  float* bnst  = alloc(512);
  __bf16* wbf  = (__bf16*)alloc(262144);   // bf16 weight tiles (1MB)
  float* fmA1  = alloc(N64 * 16);
  float* fmA2  = alloc(32768L * 32);
  float* fmA3  = alloc(4096L * 48);
  float* fmA4  = alloc(512L * 64);
  float* fmA5  = alloc(64L * 80);
  float* fp2h  = alloc(4096L * 48);
  float* mk1   = alloc(64);
  float* mk2   = alloc(4096);
  float* G0 = alloc(N64 * 32);
  float* G1 = alloc(N64 * 32);
  float* G2 = alloc(N64 * 32);
  float* G3 = alloc(N64 * 32);
  float* G4 = alloc(N64 * 32);

  // ---- launch helpers ----
  auto eblk = [](long n) { long b = (n + 255) / 256; return (int)(b > 8192 ? 8192 : b); };
  auto conv = [&](const float* in, const float* w, const float* bias, float* out,
                  int s_in, int s_out, int Cin, int Cout, int kd, int st, int pad, int mode) {
    const int nVoxT = (s_out * s_out * s_out) >> 4;
    if (Cin == 1) {
      conv_c1_kernel<<<nVoxT / 8, 256, 0, stream>>>(in, w, out);
      return;
    }
    const int taps = (mode == 1) ? 8 : kd * kd * kd;
    const int Ktot = taps * Cin;
    const int KtotPad = (Ktot + 31) & ~31;
    const int nCoutT = (Cout + 15) >> 4;
    {
      dim3 rg((KtotPad + 255) / 256, nCoutT * 16);
      repack_kernel<<<rg, 256, 0, stream>>>(w, wbf, Ktot, KtotPad, Cout);
    }
    dim3 grid((nVoxT + 7) / 8, nCoutT);
    conv_fast_kernel<<<grid, 256, (size_t)(16 * KtotPad * 2), stream>>>(
        in, wbf, bias, out, s_in, __builtin_ctz(s_in), s_out, __builtin_ctz(s_out),
        Cin, Cout, (mode == 1) ? 2 : kd, st, pad, mode, Ktot, KtotPad);
  };
  auto bn = [&](const float* x, float* y, long nvox, int C, const float* g, const float* b) {
    zero_kernel<<<1, 256, 0, stream>>>(bnst, 2L * C);
    bn_stats_kernel<<<C, 256, 0, stream>>>(x, nvox, C, bnst);
    bn_apply_kernel<<<4 * C, 256, 0, stream>>>(x, y, nvox, C, bnst, g, b);
  };
  auto resblock = [&](float* xb, float* ob, int s, int Cin, int Cout, const BlkP& p) {
    const long nvox = (long)s * s * s;
    const float* sc = xb;
    if (p.nin) { conv(xb, p.nin, nullptr, G4, s, s, Cin, Cout, 1, 1, 0, 0); sc = G4; }
    bn(xb, G1, nvox, Cin, p.g1, p.b1);
    conv(G1, p.w1, nullptr, G2, s, s, Cin, Cout, 3, 1, 1, 0);
    bn(G2, G1, nvox, Cout, p.g2, p.b2);
    conv(G1, p.w2, nullptr, G3, s, s, Cout, Cout, 3, 1, 1, 0);
    add_kernel<<<eblk(nvox * Cout), 256, 0, stream>>>(sc, G3, ob, nvox * Cout);
  };

  const int sArr[5] = {64, 32, 16, 8, 4};
  const int cArr[5] = {16, 32, 48, 64, 80};
  float* fmP[5] = {fmA1, fmA2, fmA3, fmA4, fmA5};

  // ---- input conv ----
  conv(x_in, w_in, nullptr, G0, 64, 64, 1, 16, 3, 1, 1, 0);

  // ---- encoder ----
  for (int i = 0; i < 5; ++i) {
    const int s = sArr[i], C = cArr[i];
    const long nvox = (long)s * s * s;
    resblock(G0, G0, s, C, C, enc[i].blk[0]);
    resblock(G0, G0, s, C, C, enc[i].blk[1]);
    copy_kernel<<<eblk(nvox * C), 256, 0, stream>>>(G0, fmP[i], nvox * C);
    if (i < 4) {
      bn(G0, G1, nvox, C, enc[i].dg, enc[i].db);
      conv(G1, enc[i].dw, nullptr, G0, s, s / 2, C, cArr[i + 1], 2, 2, 0, 0);
    }
  }

  // ---- decoder ----
  float* skips[4] = {fmA4, fmA3, fmA2, fmA1};
  for (int j = 0; j < 4; ++j) {
    const int lvl = 3 - j;
    const int sin = sArr[lvl + 1], sout = sArr[lvl];
    const int Cb = cArr[lvl + 1], C = cArr[lvl];
    const long nvin = (long)sin * sin * sin, nvox = (long)sout * sout * sout;
    bn(G0, G1, nvin, Cb, dec[j].g, dec[j].b);
    conv(G1, dec[j].w, nullptr, G2, sin, sout, Cb, C, 2, 1, 0, 1);   // transposed
    concat_kernel<<<4 * 2 * C, 256, 0, stream>>>(skips[j], C, G2, C, G0, nvox);
    resblock(G0, G0, sout, 2 * C, C, dec[j].blk[0]);
    resblock(G0, G0, sout, C, C, dec[j].blk[1]);
    if (j == 1) copy_kernel<<<eblk(4096L * 48), 256, 0, stream>>>(G0, fp2h, 4096L * 48);
  }

  // ---- output layout ----
  float* o_points = (float*)d_out;
  float* o_ppn1   = o_points + N64 * 5;
  float* o_ppn2   = o_ppn1 + 64L * 6;
  float* o_seg    = o_ppn2 + 4096L * 6;
  float* o_att    = o_seg + N64 * 5;
  float* o_att2   = o_att + 4096;

  // ---- seg head ----
  bn(G0, G1, N64, 16, out_g, out_b);
  conv(G1, out_wlin, out_blin, o_seg, 64, 64, 16, 5, 1, 1, 0, 0);

  // ---- PPN head chain ----
  conv(fmA5, w1c, nullptr, G2, 4, 4, 80, 80, 3, 1, 1, 0);
  conv(G2, w1s, nullptr, G3, 4, 4, 80, 2, 3, 1, 1, 0);
  ppnwrite_kernel<<<1, 256, 0, stream>>>(G3, o_ppn1, 2);
  selmask_kernel<<<1, 256, 0, stream>>>(G3, mk1, 64);
  attwrite_kernel<<<eblk(4096), 256, 0, stream>>>(mk1, 2, 2, o_att);
  maskmul_kernel<<<4 * 48, 256, 0, stream>>>(fp2h, mk1, G2, 4, 2, 48);
  conv(G2, w2c, nullptr, G4, 16, 16, 48, 48, 3, 1, 1, 0);
  conv(G4, w2s, nullptr, G3, 16, 16, 48, 2, 3, 1, 1, 0);
  ppnwrite_kernel<<<eblk(4096), 256, 0, stream>>>(G3, o_ppn2, 4);
  selmask_kernel<<<eblk(4096), 256, 0, stream>>>(G3, mk2, 4096);
  attwrite_kernel<<<eblk(N64), 256, 0, stream>>>(mk2, 4, 2, o_att2);
  maskmul_kernel<<<16 * 16, 256, 0, stream>>>(G0, mk2, G1, 6, 2, 16);
  conv(G1, w3c, nullptr, G2, 64, 64, 16, 16, 3, 1, 1, 0);
  conv(G2, w3p, nullptr, G3, 64, 64, 16, 3, 3, 1, 1, 0);
  conv(G2, w3s, nullptr, G4, 64, 64, 16, 2, 3, 1, 1, 0);
  points_kernel<<<eblk(N64), 256, 0, stream>>>(G3, G4, o_points, N64);
}
